// SANIrecursiveLayersModel_4269197492447
// MI455X (gfx1250) — matrix operand order repaired and checked by reading, not compiled, not run on hardware
//
#include <hip/hip_runtime.h>
#include <hip/hip_bf16.h>
#include <math.h>

// ---------------------------------------------------------------------------
// SANI recursive layers model, CDNA5 (gfx1250) WMMA implementation.
//
// Shapes: N=128, L=64, H=E=512, V=32000.  Output: scalar loss (1 float).
//
//   * all GEMMs via v_wmma_f32_16x16x32_f16, wave32, one wave per 16x64
//     C strip (4 N-tiles register-blocked: 4 independent accumulators share
//     one A fragment -> 4 back-to-back v_wmma with no RAW hazards)
//   * A precision (f16 vs f32-with-inline-convert) is a template parameter:
//     straight-line inner loop, no per-iteration branching
//   * W_sani split into W1/W2: per step s, P = st[0..s) @ W1^T + b_sani is one
//     big parallel GEMM; the serial chain h <- P[li] + h @ W2^T has K=512
//   * states kept f16-only (only ever a WMMA A operand)
//   * deterministic 2-stage loss reduction (no float atomics)
// ---------------------------------------------------------------------------

typedef __attribute__((ext_vector_type(16))) _Float16 v16h;
typedef __attribute__((ext_vector_type(8)))  _Float16 v8h;
typedef __attribute__((ext_vector_type(8)))  float    v8f;

#define SANI_N 128
#define SANI_L 64
#define SANI_H 512
#define SANI_E 512

// ---------------------------------------------------------------------------
// WMMA GEMM:  C[M,Nc] = A[M,K] * W[Nc,K]^T (+ bias) (+ addend)
// Grid: x = Nc/64, y = M/16.  One 32-lane wave per block (EXEC all ones).
// K multiple of 32; row strides keep 32B alignment.
// ---------------------------------------------------------------------------
template <bool AF32>
__global__ __launch_bounds__(32)
void gemm_wmma_kernel(const void*     __restrict__ Aptr,
                      long strideA,                       // elements per A row
                      const _Float16* __restrict__ W,     // [Nc, K] row-major
                      const float*    __restrict__ bias,  // [Nc] or null
                      const float*    __restrict__ addend,// [M,*] or null
                      long addStride,
                      float* __restrict__ C, long strideC,
                      int K)
{
    const int lane    = threadIdx.x & 31;
    const int l15     = lane & 15;
    const int halfsel = lane >> 4;            // 0: lanes 0-15, 1: lanes 16-31
    const int colBase = blockIdx.x * 64;
    const int tileM   = blockIdx.y;

    // A fragment: lane l15 owns row M = tileM*16 + l15.
    // ISA 16-bit A 16x32 layout: lanes 0-15 hold K {k0..k0+7, k0+16..k0+23},
    // lanes 16-31 hold K {k0+8..k0+15, k0+24..k0+31}.
    const long arow = (long)(tileM * 16 + l15) * strideA;

    // B fragments: lane l15 owns column N; same K layout as A:
    // {k0+8h..k0+8h+7, k0+16+8h..k0+16+8h+7}. W row j = B column j.
    const _Float16* wrow0 = W + (long)(colBase +  0 + l15) * K + (halfsel << 3);
    const _Float16* wrow1 = W + (long)(colBase + 16 + l15) * K + (halfsel << 3);
    const _Float16* wrow2 = W + (long)(colBase + 32 + l15) * K + (halfsel << 3);
    const _Float16* wrow3 = W + (long)(colBase + 48 + l15) * K + (halfsel << 3);

    v8f acc0 = {}, acc1 = {}, acc2 = {}, acc3 = {};

    for (int k0 = 0; k0 < K; k0 += 32) {
        v16h a;
        if constexpr (AF32) {
            const float* ar = (const float*)Aptr + arow;
            v8f f0 = *(const v8f*)(ar + k0 + (halfsel << 3));
            v8f f1 = *(const v8f*)(ar + k0 + 16 + (halfsel << 3));
            v8h h0 = __builtin_convertvector(f0, v8h);
            v8h h1 = __builtin_convertvector(f1, v8h);
            a = __builtin_shufflevector(h0, h1, 0,1,2,3,4,5,6,7,
                                               8,9,10,11,12,13,14,15);
        } else {
            const _Float16* ar = (const _Float16*)Aptr + arow;
            v8h h0 = *(const v8h*)(ar + k0 + (halfsel << 3));
            v8h h1 = *(const v8h*)(ar + k0 + 16 + (halfsel << 3));
            a = __builtin_shufflevector(h0, h1, 0,1,2,3,4,5,6,7,
                                               8,9,10,11,12,13,14,15);
        }

        v16h b0 = __builtin_shufflevector(*(const v8h*)(wrow0 + k0), *(const v8h*)(wrow0 + k0 + 16), 0,1,2,3,4,5,6,7,8,9,10,11,12,13,14,15);
        v16h b1 = __builtin_shufflevector(*(const v8h*)(wrow1 + k0), *(const v8h*)(wrow1 + k0 + 16), 0,1,2,3,4,5,6,7,8,9,10,11,12,13,14,15);
        v16h b2 = __builtin_shufflevector(*(const v8h*)(wrow2 + k0), *(const v8h*)(wrow2 + k0 + 16), 0,1,2,3,4,5,6,7,8,9,10,11,12,13,14,15);
        v16h b3 = __builtin_shufflevector(*(const v8h*)(wrow3 + k0), *(const v8h*)(wrow3 + k0 + 16), 0,1,2,3,4,5,6,7,8,9,10,11,12,13,14,15);

        // keep next K-block of the streamed B operand warm
        __builtin_prefetch(wrow0 + k0 + 32, 0, 1);

        // 4 independent accumulators -> back-to-back v_wmma, no RAW hazards
        acc0 = __builtin_amdgcn_wmma_f32_16x16x32_f16(false, a, false, b0,
                                                      (short)0, acc0, false, false);
        acc1 = __builtin_amdgcn_wmma_f32_16x16x32_f16(false, a, false, b1,
                                                      (short)0, acc1, false, false);
        acc2 = __builtin_amdgcn_wmma_f32_16x16x32_f16(false, a, false, b2,
                                                      (short)0, acc2, false, false);
        acc3 = __builtin_amdgcn_wmma_f32_16x16x32_f16(false, a, false, b3,
                                                      (short)0, acc3, false, false);
    }

    // Epilogue. C layout: VGPR i -> row (i + 8*halfsel), col = lane&15.
    const int crow0 = tileM * 16 + (halfsel << 3);
    v8f* accs[4] = { &acc0, &acc1, &acc2, &acc3 };
#pragma unroll
    for (int j = 0; j < 4; ++j) {
        const int ccol = colBase + j * 16 + l15;
        const float bv = bias ? bias[ccol] : 0.0f;
        const v8f av = *accs[j];
#pragma unroll
        for (int i = 0; i < 8; ++i) {
            const long crow = crow0 + i;
            float v = av[i] + bv;
            if (addend) v += addend[crow * addStride + ccol];
            C[crow * strideC + ccol] = v;
        }
    }
}

// ---------------------------------------------------------------------------
// Embedding gather: emb[r, :] = emb_table[labels[r], :]   (r = n*L + l)
// ---------------------------------------------------------------------------
__global__ void gather_kernel(const int* __restrict__ labels,
                              const float* __restrict__ table,
                              float* __restrict__ emb)
{
    const int r = blockIdx.x;                 // 0 .. N*L-1
    const int e = threadIdx.x * 4;            // 128 threads * 4 = 512
    const long src = (long)labels[r] * SANI_E + e;
    const long dst = (long)r * SANI_E + e;
    *(float4*)(emb + dst) = *(const float4*)(table + src);
}

// ---------------------------------------------------------------------------
// Weight conversions
// ---------------------------------------------------------------------------
__global__ void f32_to_f16_kernel(const float* __restrict__ src,
                                  _Float16* __restrict__ dst, int n)
{
    const int i = blockIdx.x * blockDim.x + threadIdx.x;
    if (i < n) dst[i] = (_Float16)src[i];
}

__global__ void split_sani_kernel(const float* __restrict__ Wsani, // [H, 2H]
                                  _Float16* __restrict__ W1,
                                  _Float16* __restrict__ W2)
{
    const int i = blockIdx.x * blockDim.x + threadIdx.x; // H*H
    if (i >= SANI_H * SANI_H) return;
    const int j = i >> 9;          // / H
    const int k = i & (SANI_H - 1);
    const long base = (long)j * (2 * SANI_H);
    W1[i] = (_Float16)Wsani[base + k];
    W2[i] = (_Float16)Wsani[base + SANI_H + k];
}

// ---------------------------------------------------------------------------
// st_h[li] = (f16) h ; optionally outs[s] = h   (src has row stride strideSrc)
// ---------------------------------------------------------------------------
__global__ void copy_state_kernel(const float* __restrict__ src, long strideSrc,
                                  _Float16* __restrict__ stDst,
                                  float* __restrict__ outsDst)
{
    const int idx = blockIdx.x * blockDim.x + threadIdx.x; // N*H
    const int n = idx >> 9;           // / H
    const int e = idx & (SANI_H - 1);
    const float v = src[(long)n * strideSrc + e];
    stDst[idx] = (_Float16)v;
    if (outsDst) outsDst[idx] = v;
}

// ---------------------------------------------------------------------------
// Loss stage 1: per (n,e) pair, log-softmax over the L axis of
// y.reshape(N,L,E) (row index n*L + l into the [L*N, E] buffer), dotted with
// emb[n,l,e]; block-reduced partials (deterministic).
// ---------------------------------------------------------------------------
__global__ void loss_partial_kernel(const float* __restrict__ y,
                                    const float* __restrict__ emb,
                                    float* __restrict__ partials)
{
    const int idx = blockIdx.x * blockDim.x + threadIdx.x; // N*E = 65536
    float part = 0.0f;
    {
        const int n = idx >> 9;          // / E
        const int e = idx & (SANI_E - 1);
        const float* yb = y   + ((long)n * SANI_L) * SANI_E + e;
        const float* eb = emb + ((long)n * SANI_L) * SANI_E + e;
        float m = -INFINITY;
        for (int l = 0; l < SANI_L; ++l)
            m = fmaxf(m, yb[(long)l * SANI_E]);
        float se = 0.0f;
        for (int l = 0; l < SANI_L; ++l)
            se += __expf(yb[(long)l * SANI_E] - m);
        const float lse = m + __logf(se);
        for (int l = 0; l < SANI_L; ++l)
            part += eb[(long)l * SANI_E] * (yb[(long)l * SANI_E] - lse);
    }
    __shared__ float sm[256];
    sm[threadIdx.x] = part;
    __syncthreads();
    for (int s = 128; s > 0; s >>= 1) {
        if ((int)threadIdx.x < s) sm[threadIdx.x] += sm[threadIdx.x + s];
        __syncthreads();
    }
    if (threadIdx.x == 0) partials[blockIdx.x] = sm[0];
}

__global__ void loss_final_kernel(const float* __restrict__ partials,
                                  float* __restrict__ out, float scale)
{
    __shared__ float sm[256];
    sm[threadIdx.x] = partials[threadIdx.x];
    __syncthreads();
    for (int s = 128; s > 0; s >>= 1) {
        if ((int)threadIdx.x < s) sm[threadIdx.x] += sm[threadIdx.x + s];
        __syncthreads();
    }
    if (threadIdx.x == 0) out[0] = -sm[0] * scale;
}

// ---------------------------------------------------------------------------
// Host side
// ---------------------------------------------------------------------------
extern "C" void kernel_launch(void* const* d_in, const int* in_sizes, int n_in,
                              void* d_out, int out_size, void* d_ws, size_t ws_size,
                              hipStream_t stream)
{
    (void)in_sizes; (void)n_in; (void)out_size; (void)ws_size;

    const int N = SANI_N, L = SANI_L, H = SANI_H, E = SANI_E;
    const long NL = (long)N * L;          // 8192
    const long NH = (long)N * H;          // 65536

    const int*   labels = (const int*)  d_in[0];
    const float* table  = (const float*)d_in[1];
    const float* W_in   = (const float*)d_in[2];
    const float* b_in   = (const float*)d_in[3];
    const float* W_sani = (const float*)d_in[4];
    const float* b_sani = (const float*)d_in[5];
    const float* W_out  = (const float*)d_in[6];
    const float* b_out  = (const float*)d_in[7];
    float* out = (float*)d_out;

    // Workspace carving (all offsets keep >=256B alignment). ~79 MB total.
    char* ws = (char*)d_ws;
    size_t off = 0;
    auto carve = [&](size_t bytes) { char* p = ws + off; off += (bytes + 255) & ~(size_t)255; return p; };

    float*    emb    = (float*)   carve(NL * E * sizeof(float));      // 16.8 MB
    float*    x      = (float*)   carve(NL * H * sizeof(float));      // 16.8 MB
    _Float16* st_h   = (_Float16*)carve((size_t)L * NH * 2);          //  8.4 MB
    float*    P      = (float*)   carve((size_t)L * NH * 4);          // 16.8 MB (aliased as y later)
    float*    outs   = (float*)   carve((size_t)L * NH * 4);          // 16.8 MB
    float*    hA     = (float*)   carve(NH * 4);                      //  0.26 MB
    float*    hB     = (float*)   carve(NH * 4);                      //  0.26 MB
    _Float16* W_in_h = (_Float16*)carve((size_t)H * E * 2);
    _Float16* W1_h   = (_Float16*)carve((size_t)H * H * 2);
    _Float16* W2_h   = (_Float16*)carve((size_t)H * H * 2);
    _Float16* W_out_h= (_Float16*)carve((size_t)E * H * 2);
    float*    parts  = (float*)   carve(256 * sizeof(float));
    float*    y      = P;   // P is dead once the recurrence finishes

    // 1) gather + weight conversion
    gather_kernel<<<dim3((unsigned)NL), dim3(128), 0, stream>>>(labels, table, emb);
    f32_to_f16_kernel<<<dim3((H * E + 255) / 256), dim3(256), 0, stream>>>(W_in, W_in_h, H * E);
    f32_to_f16_kernel<<<dim3((E * H + 255) / 256), dim3(256), 0, stream>>>(W_out, W_out_h, E * H);
    split_sani_kernel<<<dim3((H * H + 255) / 256), dim3(256), 0, stream>>>(W_sani, W1_h, W2_h);

    // 2) x = emb @ W_in^T + b_in   ([8192,512] x [512,512])
    gemm_wmma_kernel<true><<<dim3(H / 64, (unsigned)(NL / 16)), dim3(32), 0, stream>>>(
        emb, (long)E, W_in_h, b_in, nullptr, 0, x, (long)H, E);

    // 3) recurrence
    for (int s = 0; s < L; ++s) {
        if (s > 0) {
            // P[0..s) = st @ W1^T + b_sani   (parallel across all li < s)
            gemm_wmma_kernel<false><<<dim3(H / 64, (unsigned)((long)s * N / 16)), dim3(32), 0, stream>>>(
                st_h, (long)H, W1_h, b_sani, nullptr, 0, P, (long)H, H);
        }
        const float* hsrc = x + (long)s * H;      // h0 = x[:, s, :]
        long hstride = (long)L * H;
        int pp = 0;
        for (int li = 0; li < s; ++li) {
            // st[li] = h_li (f16)
            copy_state_kernel<<<dim3((unsigned)(NH / 256)), dim3(256), 0, stream>>>(
                hsrc, hstride, st_h + (long)li * NH, nullptr);
            // h <- P[li] + h @ W2^T   (serial chain, K=512)
            float* hdst = (pp == 0) ? hA : hB;
            gemm_wmma_kernel<true><<<dim3(H / 64, N / 16), dim3(32), 0, stream>>>(
                hsrc, hstride, W2_h, nullptr,
                P + (long)li * NH, (long)H, hdst, (long)H, H);
            hsrc = hdst; hstride = (long)H; pp ^= 1;
        }
        // st[s] = h_s ; outs[s] = h_s
        copy_state_kernel<<<dim3((unsigned)(NH / 256)), dim3(256), 0, stream>>>(
            hsrc, hstride, st_h + (long)s * NH, outs + (long)s * NH);
    }

    // 4) y = outs.reshape(L*N, H) @ W_out^T + b_out
    gemm_wmma_kernel<true><<<dim3(E / 64, (unsigned)(NL / 16)), dim3(32), 0, stream>>>(
        outs, (long)H, W_out_h, b_out, nullptr, 0, y, (long)E, H);

    // 5) loss = -(emb * log_softmax(y, axis=L)).sum(L).mean()
    loss_partial_kernel<<<dim3(256), dim3(256), 0, stream>>>(y, emb, parts);
    loss_final_kernel<<<dim3(1), dim3(256), 0, stream>>>(parts, out, 1.0f / (float)(N * E));
}
